// GraphAttentionLayer_5858335392466
// MI455X (gfx1250) — compile-verified, matrix operand order checked by reading, not run on hardware
//
#include <hip/hip_runtime.h>
#include <hip/hip_bf16.h>

#define N_NODES 8192
#define IN_F    256
#define OUT_F   128
#define NEG_SLOPE 0.2f

typedef __attribute__((ext_vector_type(16))) __bf16 v16bf;
typedef __attribute__((ext_vector_type(8)))  float  v8f;
typedef __attribute__((ext_vector_type(2)))  float  v2f;
typedef __attribute__((ext_vector_type(4)))  int    v4i;

// Async global->LDS staging path (CDNA5 ASYNCcnt). Guarded so the file
// compiles on toolchains lacking the builtins (falls back to direct loads).
#if defined(__AMDGCN__) &&                                                  \
    __has_builtin(__builtin_amdgcn_global_load_async_to_lds_b128) &&        \
    __has_builtin(__builtin_amdgcn_s_wait_asynccnt)
#define GAT_ASYNC 1
#else
#define GAT_ASYNC 0
#endif

#define TILE_STRIDE 36                  // padded LDS row stride (dwords): 36*lm mod 64 hits 16 distinct banks
#define TILE_SZ     (16 * TILE_STRIDE)  // 576 dwords per A-tile buffer

// ---------------------------------------------------------------------------
// Kernel 1: Z = X @ W via v_wmma_f32_16x16x4_f32 (full f32 precision).
// Also emits Zt = bf16 transpose of Z for the pass-2 WMMA B operand.
// ---------------------------------------------------------------------------
__global__ __launch_bounds__(256) void zgemm_kernel(
    const float* __restrict__ X, const float* __restrict__ W,
    float* __restrict__ Z, __bf16* __restrict__ Zt)
{
    const int tid   = threadIdx.x;
    const int lane  = tid & 31;
    const int half  = lane >> 4;
    const int lm    = lane & 15;
    const int wave  = tid >> 5;
    const int ibase = blockIdx.x * 16;
    const int fbase = wave * 16;

    v8f acc = {0.f,0.f,0.f,0.f,0.f,0.f,0.f,0.f};

    const int row = ibase + lm;
    const int n   = fbase + lm;
    #pragma unroll 4
    for (int k = 0; k < IN_F; k += 4) {
        v2f a2 = *(const v2f*)(X + (size_t)row * IN_F + k + 2 * half);
        v2f b2;
        b2[0] = W[(size_t)(k + 2 * half)     * OUT_F + n];
        b2[1] = W[(size_t)(k + 2 * half + 1) * OUT_F + n];
        acc = __builtin_amdgcn_wmma_f32_16x16x4_f32(
            false, a2, false, b2, (short)0, acc, false, false);
    }

    #pragma unroll
    for (int v = 0; v < 8; ++v) {
        const int r = ibase + v + 8 * half;
        const float z = acc[v];
        Z [(size_t)r * OUT_F + n] = z;
        Zt[(size_t)n * N_NODES + r] = (__bf16)z;
    }
}

// ---------------------------------------------------------------------------
// Kernel 2: s_src = Z @ a[:128], s_dst = Z @ a[128:].  One wave per row.
// ---------------------------------------------------------------------------
__global__ __launch_bounds__(256) void svec_kernel(
    const float* __restrict__ Z, const float* __restrict__ avec,
    float* __restrict__ s_src, float* __restrict__ s_dst)
{
    const int lane = threadIdx.x & 31;
    const int row  = blockIdx.x * 8 + (threadIdx.x >> 5);

    const float4 z  = *(const float4*)(Z    + (size_t)row * OUT_F + lane * 4);
    const float4 as = *(const float4*)(avec + lane * 4);
    const float4 ad = *(const float4*)(avec + OUT_F + lane * 4);

    float ss = z.x * as.x + z.y * as.y + z.z * as.z + z.w * as.w;
    float sd = z.x * ad.x + z.y * ad.y + z.z * ad.z + z.w * ad.w;
    #pragma unroll
    for (int off = 16; off > 0; off >>= 1) {
        ss += __shfl_xor(ss, off, 32);
        sd += __shfl_xor(sd, off, 32);
    }
    if (lane == 0) { s_src[row] = ss; s_dst[row] = sd; }
}

// ---------------------------------------------------------------------------
// Kernel 3: per-row softmax stats over ALL j (non-edges contribute exp(-m)).
// Fully coalesced b128 streaming of the A row.
// ---------------------------------------------------------------------------
__global__ __launch_bounds__(256) void stats_kernel(
    const int* __restrict__ A,
    const float* __restrict__ s_src, const float* __restrict__ s_dst,
    float* __restrict__ Mrow, float* __restrict__ Linv)
{
    __shared__ float red[256];
    const int tid = threadIdx.x;
    const int row = blockIdx.x;
    const int* Ar = A + ((long long)row << 13);
    const float si = s_src[row];

    float e[32];
    float m = -1e30f;
    #pragma unroll
    for (int q = 0; q < 8; ++q) {
        const int col = (q << 10) + tid * 4;          // 8 x int4 per thread
        const int4   a4 = *(const int4*)  (Ar    + col);
        const float4 s4 = *(const float4*)(s_dst + col);
        const float r0 = si + s4.x, r1 = si + s4.y, r2 = si + s4.z, r3 = si + s4.w;
        e[4*q+0] = (a4.x != 0) ? fmaxf(r0, NEG_SLOPE * r0) : 0.f;
        e[4*q+1] = (a4.y != 0) ? fmaxf(r1, NEG_SLOPE * r1) : 0.f;
        e[4*q+2] = (a4.z != 0) ? fmaxf(r2, NEG_SLOPE * r2) : 0.f;
        e[4*q+3] = (a4.w != 0) ? fmaxf(r3, NEG_SLOPE * r3) : 0.f;
        m = fmaxf(m, fmaxf(fmaxf(e[4*q], e[4*q+1]), fmaxf(e[4*q+2], e[4*q+3])));
    }
    red[tid] = m; __syncthreads();
    #pragma unroll
    for (int s = 128; s > 0; s >>= 1) {
        if (tid < s) red[tid] = fmaxf(red[tid], red[tid + s]);
        __syncthreads();
    }
    m = red[0]; __syncthreads();

    float sum = 0.f;
    #pragma unroll
    for (int q = 0; q < 32; ++q) sum += __expf(e[q] - m);
    red[tid] = sum; __syncthreads();
    #pragma unroll
    for (int s = 128; s > 0; s >>= 1) {
        if (tid < s) red[tid] += red[tid + s];
        __syncthreads();
    }
    if (tid == 0) { Mrow[row] = m; Linv[row] = 1.f / red[0]; }
}

// ---------------------------------------------------------------------------
// Kernel 4: out = softmax(e) @ Z via bf16 WMMA.
// Block = 16-row stripe; 8 waves split the j range; A tiles staged to LDS
// with double-buffered GLOBAL_LOAD_ASYNC_TO_LDS_B128 (fully coalesced rows),
// pipelined on ASYNCcnt. LDS cross-wave split-K reduction at the end.
// ---------------------------------------------------------------------------
__global__ __launch_bounds__(256) void gat_kernel(
    const int* __restrict__ A,
    const float* __restrict__ s_src, const float* __restrict__ s_dst,
    const float* __restrict__ Mrow, const float* __restrict__ Linv,
    const __bf16* __restrict__ Zt, float* __restrict__ out)
{
#if GAT_ASYNC
    __shared__ int   atile[8 * 2 * TILE_SZ];   // 36 KB: wave-private double buffers
#endif
    __shared__ float redsh[8 * 256];           //  8 KB: split-K reduction

    const int tid   = threadIdx.x;
    const int lane  = tid & 31;
    const int half  = lane >> 4;
    const int lm    = lane & 15;
    const int wave  = tid >> 5;
    const int ibase = blockIdx.x * 16;

    const int  row = ibase + lm;               // P-gen row for this lane
    const float si = s_src[row];
    const float mr = Mrow[row];
    const float em = __expf(-mr);              // non-edge contribution

    v8f acc[8];
    #pragma unroll
    for (int t = 0; t < 8; ++t)
        acc[t] = (v8f){0.f,0.f,0.f,0.f,0.f,0.f,0.f,0.f};

    const int jbeg = wave * (N_NODES / 8);
    const int jend = jbeg + (N_NODES / 8);

#if GAT_ASYNC
    int* buf0 = atile + wave * 2 * TILE_SZ;
    int* buf1 = buf0 + TILE_SZ;
    const int fr = lane >> 3;                  // row within quad of rows
    const int fc = (lane & 7) * 4;             // dword column
    auto fetch = [&](int j, int* buf) {
        #pragma unroll
        for (int q = 0; q < 4; ++q) {          // 4 async b128 ops = 16 full rows
            const int r = 4 * q + fr;
            const int* g = A + (size_t)(ibase + r) * N_NODES + j + fc;
            __builtin_amdgcn_global_load_async_to_lds_b128(
                (__attribute__((address_space(1))) v4i*)g,
                (__attribute__((address_space(3))) v4i*)(buf + r * TILE_STRIDE + fc),
                0, 0);
        }
    };
    fetch(jbeg, buf0);
#else
    const int* Ar = A + ((long long)row << 13);
#endif

    for (int j = jbeg; j < jend; j += 32) {
        union { int4 q[4]; int e[16]; } aa;
#if GAT_ASYNC
        int* cur = (((j - jbeg) >> 5) & 1) ? buf1 : buf0;
        int* nxt = (((j - jbeg) >> 5) & 1) ? buf0 : buf1;
        if (j + 32 < jend) {
            fetch(j + 32, nxt);
            __builtin_amdgcn_s_wait_asynccnt(4);   // current tile landed in LDS
        } else {
            __builtin_amdgcn_s_wait_asynccnt(0);
        }
        const int* rowp = cur + lm * TILE_STRIDE;  // conflict-free ds reads
        aa.q[0] = *(const int4*)(rowp +      8 * half);
        aa.q[1] = *(const int4*)(rowp +  4 + 8 * half);
        aa.q[2] = *(const int4*)(rowp + 16 + 8 * half);
        aa.q[3] = *(const int4*)(rowp + 20 + 8 * half);
#else
        aa.q[0] = *(const int4*)(Ar + j +      8 * half);
        aa.q[1] = *(const int4*)(Ar + j +  4 + 8 * half);
        aa.q[2] = *(const int4*)(Ar + j + 16 + 8 * half);
        aa.q[3] = *(const int4*)(Ar + j + 20 + 8 * half);
        if (j + 32 < jend)
            __builtin_prefetch(Ar + j + 32 + 8 * half, 0, 1);
#endif

        // Build P tile directly in the WMMA A-operand layout
        union { v16bf v; __bf16 e[16]; } af;
        #pragma unroll
        for (int v = 0; v < 8; ++v) {
            const int kbase = ((v >> 2) << 4) + (half << 3) + ((v & 3) << 1);
            const int eidx  = ((v >> 2) << 3) + ((v & 3) << 1);
            const float2 sd2 = *(const float2*)(s_dst + j + kbase); // half-wave uniform, L2 hit
            float r0 = si + sd2.x, r1 = si + sd2.y;
            float l0 = fmaxf(r0, NEG_SLOPE * r0);
            float l1 = fmaxf(r1, NEG_SLOPE * r1);
            float p0 = (aa.e[eidx]     != 0) ? __expf(l0 - mr) : em;
            float p1 = (aa.e[eidx + 1] != 0) ? __expf(l1 - mr) : em;
            af.e[2 * v]     = (__bf16)p0;
            af.e[2 * v + 1] = (__bf16)p1;
        }

        // 8 f-tiles: acc[t] += P (16x32) @ Zt-tile (32x16)
        #pragma unroll
        for (int t = 0; t < 8; ++t) {
            union { v16bf v; uint4 q[2]; } bf;
            const __bf16* zr = Zt + (size_t)(t * 16 + lm) * N_NODES + j;
            bf.q[0] = *(const uint4*)(zr +      8 * half);
            bf.q[1] = *(const uint4*)(zr + 16 + 8 * half);
            acc[t] = __builtin_amdgcn_wmma_f32_16x16x32_bf16(
                false, af.v, false, bf.v, (short)0, acc[t], false, false);
        }
    }

    __syncthreads();
    // Split-K reduction across the 8 waves, one f-tile at a time
    for (int t = 0; t < 8; ++t) {
        #pragma unroll
        for (int v = 0; v < 8; ++v)
            redsh[wave * 256 + v * 32 + lane] = acc[t][v];
        __syncthreads();
        float val = 0.f;
        #pragma unroll
        for (int w = 0; w < 8; ++w) val += redsh[w * 256 + tid];
        const int vv = tid >> 5, l = tid & 31;
        const int r = ibase + vv + 8 * (l >> 4);
        const int c = t * 16 + (l & 15);
        out[(size_t)r * OUT_F + c] = val * Linv[r];
        __syncthreads();
    }
}

// ---------------------------------------------------------------------------
extern "C" void kernel_launch(void* const* d_in, const int* in_sizes, int n_in,
                              void* d_out, int out_size, void* d_ws, size_t ws_size,
                              hipStream_t stream) {
    const float* X    = (const float*)d_in[0];   // 8192 x 256
    const int*   A    = (const int*)  d_in[1];   // 8192 x 8192
    const float* W    = (const float*)d_in[2];   // 256 x 128
    const float* avec = (const float*)d_in[3];   // 256
    float* out = (float*)d_out;                  // 8192 x 128 f32

    char* ws = (char*)d_ws;
    float*  Z     = (float*)ws;                          // 4 MB
    __bf16* Zt    = (__bf16*)(ws + (4u << 20));          // 2 MB
    float*  s_src = (float*)(ws + (6u << 20));           // 32 KB
    float*  s_dst = s_src + N_NODES;                     // 32 KB
    float*  Mrow  = s_dst + N_NODES;                     // 32 KB
    float*  Linv  = Mrow  + N_NODES;                     // 32 KB

    zgemm_kernel<<<N_NODES / 16, 256, 0, stream>>>(X, W, Z, Zt);
    svec_kernel <<<N_NODES / 8,  256, 0, stream>>>(Z, avec, s_src, s_dst);
    stats_kernel<<<N_NODES,      256, 0, stream>>>(A, s_src, s_dst, Mrow, Linv);
    gat_kernel  <<<N_NODES / 16, 256, 0, stream>>>(A, s_src, s_dst, Mrow, Linv, Zt, out);
}
